// FDModel_87651692577337
// MI455X (gfx1250) — matrix operand order repaired and checked by reading, not compiled, not run on hardware
//
#include <hip/hip_runtime.h>
#include <hip/hip_bf16.h>

// MI455X / gfx1250 implementation.
// Precision: f16 WMMA operands (v_wmma_f32_16x16x32_f16), f32 accumulation.
// Fused attention (two-pass, no V*N*N materialization).
// Workspace requirement: ~37 MB (h f32, q f16, masked-transposed h f16, row sums).

#define V_ 3
#define N_ 6000
#define D_ 512
#define H_ 256
#define DY_ 512
#define SLOPE_ 0.1f
#define INV_BETA_ 5.0f   // 1/0.2

typedef _Float16 f16;
typedef __attribute__((ext_vector_type(16))) _Float16 v16h;
typedef __attribute__((ext_vector_type(8)))  _Float16 v8h;
typedef __attribute__((ext_vector_type(8)))  float    v8f;

__device__ __forceinline__ v16h cat16(v8h lo, v8h hi) {
  v16h r;
#pragma unroll
  for (int t = 0; t < 8; ++t) { r[t] = lo[t]; r[8 + t] = hi[t]; }
  return r;
}

__device__ __forceinline__ void pack4(v16h& d, int o, float4 f) {
  d[o + 0] = (f16)f.x; d[o + 1] = (f16)f.y; d[o + 2] = (f16)f.z; d[o + 3] = (f16)f.w;
}

// ---------------------------------------------------------------------------
// Generic C = epi(A(MxK) * B(NcxK)^T + bias), both operands row-major and
// K-contiguous -> per-lane contiguous b128 operand loads, one 16x16 C tile
// per wave, f32->f16 conversion on the fly, f32 accumulation.
// ---------------------------------------------------------------------------
enum { EPI_LEAKY = 0, EPI_LEAKY_NAN = 1, EPI_SIGMOID = 2 };

template <int EPI>
__global__ __launch_bounds__(128)
void gemm_nt_wmma(const float* __restrict__ A, const float* __restrict__ Bm,
                  const float* __restrict__ bias, float* __restrict__ C,
                  int M, int Nc, int K,
                  long sA, long sB, long sBias, long sC) {
  const int v = blockIdx.z;
  A += (long)v * sA; Bm += (long)v * sB; bias += (long)v * sBias; C += (long)v * sC;
  const int lane = threadIdx.x & 31;
  const int wave = threadIdx.x >> 5;
  const int half = lane >> 4;
  const int l16  = lane & 15;
  const int tn = blockIdx.x;
  const int tm = blockIdx.y * 4 + wave;
  if (tm * 16 >= M) return;

  const float* arow = A  + (long)(tm * 16 + l16) * K;   // A lane row = M (=l16)
  const float* brow = Bm + (long)(tn * 16 + l16) * K;   // B lane col = N (=l16)
  v8f acc = {};
  for (int k = 0; k < K; k += 32) {
    if (k + 64 < K) {
      __builtin_prefetch(arow + k + 64, 0, 1);
      __builtin_prefetch(brow + k + 64, 0, 1);
    }
    // A (16x32 f16): lanes 0-15 hold K {0..7,16..23}; lanes 16-31 K {8..15,24..31}
    float4 a0 = *(const float4*)(arow + k + 8 * half);
    float4 a1 = *(const float4*)(arow + k + 8 * half + 4);
    float4 a2 = *(const float4*)(arow + k + 16 + 8 * half);
    float4 a3 = *(const float4*)(arow + k + 16 + 8 * half + 4);
    // B (32x16 f16): lanes 0-15 hold K {0..7,16..23}; lanes 16-31 K {8..15,24..31}
    float4 b0 = *(const float4*)(brow + k + 8 * half);
    float4 b1 = *(const float4*)(brow + k + 8 * half + 4);
    float4 b2 = *(const float4*)(brow + k + 16 + 8 * half);
    float4 b3 = *(const float4*)(brow + k + 16 + 8 * half + 4);
    v16h a, b;
    pack4(a, 0, a0); pack4(a, 4, a1); pack4(a, 8, a2); pack4(a, 12, a3);
    pack4(b, 0, b0); pack4(b, 4, b1); pack4(b, 8, b2); pack4(b, 12, b3);
    acc = __builtin_amdgcn_wmma_f32_16x16x32_f16(false, a, false, b,
                                                 (short)0, acc, false, false);
  }
  const float bs = bias[tn * 16 + l16];
#pragma unroll
  for (int r = 0; r < 8; ++r) {
    float x = acc[r] + bs;   // C element (M = r + 8*half, N = l16)
    if (EPI == EPI_SIGMOID) {
      x = 1.f / (1.f + __expf(-x));
    } else {
      x = (x >= 0.f) ? x : SLOPE_ * x;
      if (EPI == EPI_LEAKY_NAN) {
        if (x != x) x = 0.f;
        else if (x >  3.402823466e+38f) x =  3.402823466e+38f;
        else if (x < -3.402823466e+38f) x = -3.402823466e+38f;
      }
    }
    C[(long)(tm * 16 + r + 8 * half) * Nc + tn * 16 + l16] = x;
  }
}

// ---------------------------------------------------------------------------
// q = h / max(||h||, 1e-12) (f16), and hTm[v][h][n] = h * mT[v][n] (f16,
// transposed + column-mask folded in, so pass-2 B operands are contiguous).
// One wave per row (H=256 -> 8 elems/lane).
// ---------------------------------------------------------------------------
__global__ __launch_bounds__(256)
void rownorm_kernel(const float* __restrict__ h, const float* __restrict__ mask,
                    f16* __restrict__ q, f16* __restrict__ hTm) {
  const int lane = threadIdx.x & 31;
  const int wave = threadIdx.x >> 5;
  const long g = (long)blockIdx.x * 8 + wave;  // 0 .. V*N-1
  const int v = (int)(g / N_);
  const int n = (int)(g % N_);
  const float* row = h + g * H_;
  float vals[8];
  float ss = 0.f;
#pragma unroll
  for (int t = 0; t < 8; ++t) { vals[t] = row[lane + 32 * t]; ss += vals[t] * vals[t]; }
#pragma unroll
  for (int o = 16; o > 0; o >>= 1) ss += __shfl_xor(ss, o, 32);
  const float inv = 1.f / fmaxf(sqrtf(ss), 1e-12f);
  const float mv = mask[(long)n * V_ + v];
#pragma unroll
  for (int t = 0; t < 8; ++t) {
    const int hc = lane + 32 * t;
    q[g * H_ + hc] = (f16)(vals[t] * inv);
    hTm[((long)v * H_ + hc) * N_ + n] = (f16)(vals[t] * mv);
  }
}

// ---------------------------------------------------------------------------
// Pass 1: S[n] = sum_m A[n,m], A = max_v exp(5*q_v[n].q_v[m]) * m_v[n]*m_v[m],
// diag zeroed. One block per 16-row n-tile; 8 waves stripe the m tiles;
// scores recomputed by WMMA; row sums via LDS ds_add_f32.
// ---------------------------------------------------------------------------
__global__ __launch_bounds__(256)
void att_rowsum_kernel(const f16* __restrict__ q, const float* __restrict__ mask,
                       float* __restrict__ S) {
  __shared__ float rowsum[16];
  __shared__ float mrow[3][16];
  const int tid = threadIdx.x;
  const int lane = tid & 31, wave = tid >> 5;
  const int half = lane >> 4, l16 = lane & 15;
  const int n0 = blockIdx.x * 16;
  if (tid < 16) rowsum[tid] = 0.f;
  if (tid < 48) mrow[tid >> 4][tid & 15] = mask[(long)(n0 + (tid & 15)) * V_ + (tid >> 4)];
  __syncthreads();

  for (int mt = wave; mt < N_ / 16; mt += 8) {
    const int m0 = mt * 16;
    float amax[8];
#pragma unroll
    for (int r = 0; r < 8; ++r) amax[r] = 0.f;
    for (int v = 0; v < V_; ++v) {
      const f16* qa = q + ((long)v * N_ + n0 + l16) * H_;
      const f16* qb = q + ((long)v * N_ + m0 + l16) * H_;
      v8f acc = {};
#pragma unroll
      for (int kc = 0; kc < 8; ++kc) {
        const int k = kc * 32;
        v8h alo = *(const v8h*)(qa + k + 8 * half);
        v8h ahi = *(const v8h*)(qa + k + 16 + 8 * half);
        v8h blo = *(const v8h*)(qb + k + 8 * half);
        v8h bhi = *(const v8h*)(qb + k + 16 + 8 * half);
        acc = __builtin_amdgcn_wmma_f32_16x16x32_f16(false, cat16(alo, ahi),
                                                     false, cat16(blo, bhi),
                                                     (short)0, acc, false, false);
      }
      const float mcol = mask[(long)(m0 + l16) * V_ + v];
#pragma unroll
      for (int r = 0; r < 8; ++r) {
        const float a = __expf(acc[r] * INV_BETA_) * mrow[v][r + 8 * half] * mcol;
        amax[r] = fmaxf(amax[r], a);
      }
    }
#pragma unroll
    for (int r = 0; r < 8; ++r) {
      if (n0 + r + 8 * half == m0 + l16) amax[r] = 0.f;  // zero diagonal
      atomicAdd(&rowsum[r + 8 * half], amax[r]);         // ds_add_f32
    }
  }
  __syncthreads();
  if (tid < 16) S[n0 + tid] = rowsum[tid];
}

// ---------------------------------------------------------------------------
// Pass 2: new_x[v,n,:] = (1-m_v[n]) * sum_m (A[n,m]/(S+1e-9)) * hTm_v[:,m]
//                        + m_v[n] * h[v,n,:]
// One block per 16-row n-tile. Per 32-wide m chunk:
//   stage1: waves 0-5 compute the 6 (v, col-half) exp score tiles -> LDS
//   stage2: all threads reduce max over v, zero diag, /S -> f16 P tile in LDS
//   stage3: all 8 waves run P @ hTm for their 6 (v, h-tile) accumulators
// ---------------------------------------------------------------------------
__global__ __launch_bounds__(256)
void att_apply_kernel(const f16* __restrict__ q, const f16* __restrict__ hTm,
                      const float* __restrict__ h, const float* __restrict__ mask,
                      const float* __restrict__ S, float* __restrict__ newx) {
  __shared__ float sc[3][16][32];
  __shared__ __align__(16) f16 P[16][32];
  __shared__ float Sl[16];
  __shared__ float mrow[3][16];
  const int tid = threadIdx.x;
  const int lane = tid & 31, wave = tid >> 5;
  const int half = lane >> 4, l16 = lane & 15;
  const int n0 = blockIdx.x * 16;
  if (tid < 16) Sl[tid] = S[n0 + tid] + 1e-9f;
  if (tid < 48) mrow[tid >> 4][tid & 15] = mask[(long)(n0 + (tid & 15)) * V_ + (tid >> 4)];
  __syncthreads();

  v8f acc[6] = {};   // 48 tasks = 3 views x 16 h-tiles, 6 per wave

  for (int m0 = 0; m0 < N_; m0 += 32) {
    // ---- stage 1: masked exp scores for this 16n x 32m chunk -> LDS
    if (wave < 6) {
      const int v  = wave >> 1;
      const int ch = wave & 1;
      const int mb = m0 + ch * 16;
      float vals[8];
      if (mb < N_) {
        const f16* qa = q + ((long)v * N_ + n0 + l16) * H_;
        const f16* qb = q + ((long)v * N_ + mb + l16) * H_;
        v8f s = {};
#pragma unroll
        for (int kc = 0; kc < 8; ++kc) {
          const int k = kc * 32;
          v8h alo = *(const v8h*)(qa + k + 8 * half);
          v8h ahi = *(const v8h*)(qa + k + 16 + 8 * half);
          v8h blo = *(const v8h*)(qb + k + 8 * half);
          v8h bhi = *(const v8h*)(qb + k + 16 + 8 * half);
          s = __builtin_amdgcn_wmma_f32_16x16x32_f16(false, cat16(alo, ahi),
                                                     false, cat16(blo, bhi),
                                                     (short)0, s, false, false);
        }
        const float mcol = mask[(long)(mb + l16) * V_ + v];
#pragma unroll
        for (int r = 0; r < 8; ++r)
          vals[r] = __expf(s[r] * INV_BETA_) * mrow[v][r + 8 * half] * mcol;
      } else {
#pragma unroll
        for (int r = 0; r < 8; ++r) vals[r] = 0.f;
      }
#pragma unroll
      for (int r = 0; r < 8; ++r)
        sc[v][r + 8 * half][ch * 16 + l16] = vals[r];
    }
    __syncthreads();

    // ---- stage 2: A = max over views, zero diag, normalize, -> f16 P
    for (int e = tid; e < 512; e += 256) {
      const int nl = e >> 5, ml = e & 31;
      float a = fmaxf(sc[0][nl][ml], fmaxf(sc[1][nl][ml], sc[2][nl][ml]));
      const int gm = m0 + ml;
      if (gm == n0 + nl || gm >= N_) a = 0.f;
      P[nl][ml] = (f16)(a / Sl[nl]);
    }
    __syncthreads();

    // ---- stage 3: accumulate P (16x32) @ hTm (32 x 16 per tile)
    v8h plo = *(const v8h*)&P[l16][16 * half];
    v8h phi = *(const v8h*)&P[l16][16 * half + 8];
    const v16h pa = cat16(plo, phi);
    int mb2 = m0 + 16 * half;
    if (mb2 >= N_) mb2 = 0;  // those P columns are zero; clamp keeps loads in-bounds
#pragma unroll
    for (int i = 0; i < 6; ++i) {
      const int task = wave * 6 + i;
      const int v = task >> 4, ht = task & 15;
      const f16* bp = hTm + ((long)v * H_ + ht * 16 + l16) * N_ + mb2;
      v8h blo = *(const v8h*)bp;
      v8h bhi = *(const v8h*)(bp + 8);
      acc[i] = __builtin_amdgcn_wmma_f32_16x16x32_f16(false, pa, false, cat16(blo, bhi),
                                                      (short)0, acc[i], false, false);
    }
    __syncthreads();  // protect sc/P before next chunk overwrites
  }

  // ---- epilogue: blend with h by row mask, store new_x
#pragma unroll
  for (int i = 0; i < 6; ++i) {
    const int task = wave * 6 + i;
    const int v = task >> 4, ht = task & 15;
    const int col = ht * 16 + l16;
#pragma unroll
    for (int r = 0; r < 8; ++r) {
      const int n = n0 + r + 8 * half;
      const float mv = mrow[v][r + 8 * half];
      const float hv = h[((long)v * N_ + n) * H_ + col];
      newx[((long)v * N_ + n) * H_ + col] = acc[i][r] * (1.f - mv) + hv * mv;
    }
  }
}

// ---------------------------------------------------------------------------
extern "C" void kernel_launch(void* const* d_in, const int* in_sizes, int n_in,
                              void* d_out, int out_size, void* d_ws, size_t ws_size,
                              hipStream_t stream) {
  const float* x     = (const float*)d_in[0];
  const float* y     = (const float*)d_in[1];
  const float* mask  = (const float*)d_in[2];
  const float* W_in  = (const float*)d_in[3];
  const float* b_in  = (const float*)d_in[4];
  const float* W_out = (const float*)d_in[5];
  const float* b_out = (const float*)d_in[6];
  const float* Wy    = (const float*)d_in[7];
  const float* by    = (const float*)d_in[8];

  float* out_newx = (float*)d_out;                        // V*N*H
  float* out_xnew = out_newx + (size_t)V_ * N_ * H_;      // V*N*D
  float* out_yn   = out_xnew + (size_t)V_ * N_ * D_;      // N*H

  char* ws = (char*)d_ws;
  const size_t hB = (size_t)V_ * N_ * H_;                 // element counts
  float* h_f32 = (float*)ws;                              // hB * 4 bytes
  f16*   q_f16 = (f16*)(ws + hB * 4);                     // hB * 2 bytes
  f16*   hTm   = (f16*)(ws + hB * 4 + hB * 2);            // hB * 2 bytes
  float* S     = (float*)(ws + hB * 4 + hB * 4);          // N_ * 4 bytes

  const dim3 blk(128);
  const int mblk = (N_ / 16 + 3) / 4;                     // 4 waves / block along M

  // h = leaky(x @ W_in^T + b_in)   (per-view GEMM, K = D)
  gemm_nt_wmma<EPI_LEAKY><<<dim3(H_ / 16, mblk, V_), blk, 0, stream>>>(
      x, W_in, b_in, h_f32, N_, H_, D_,
      (long)N_ * D_, (long)H_ * D_, (long)H_, (long)N_ * H_);

  // q (unit rows, f16) and masked-transposed h (f16)
  rownorm_kernel<<<dim3(V_ * N_ / 8), dim3(256), 0, stream>>>(h_f32, mask, q_f16, hTm);

  // pass 1: attention row sums
  att_rowsum_kernel<<<dim3(N_ / 16), dim3(256), 0, stream>>>(q_f16, mask, S);

  // pass 2: new_x = blend(A/S @ (h * colmask), h)
  att_apply_kernel<<<dim3(N_ / 16), dim3(256), 0, stream>>>(
      q_f16, hTm, h_f32, mask, S, out_newx);

  // x_new = nan_to_num(leaky(new_x @ W_out^T + b_out))   (K = H)
  gemm_nt_wmma<EPI_LEAKY_NAN><<<dim3(D_ / 16, mblk, V_), blk, 0, stream>>>(
      out_newx, W_out, b_out, out_xnew, N_, D_, H_,
      (long)N_ * H_, (long)D_ * H_, (long)D_, (long)N_ * D_);

  // y_n = sigmoid(y @ Wy^T + by)   (K = DY)
  gemm_nt_wmma<EPI_SIGMOID><<<dim3(H_ / 16, mblk, 1), blk, 0, stream>>>(
      y, Wy, by, out_yn, N_, H_, DY_, 0L, 0L, 0L, 0L);
}
